// MultiHeadAttention_17325898072778
// MI455X (gfx1250) — matrix operand order repaired and checked by reading, not compiled, not run on hardware
//
#include <hip/hip_runtime.h>
#include <hip/hip_bf16.h>
#include <math.h>

typedef __bf16 bf16;
typedef __attribute__((ext_vector_type(16))) __bf16 v16bf;
typedef __attribute__((ext_vector_type(8)))  __bf16 v8bf;
typedef __attribute__((ext_vector_type(8)))  float  v8f;

#define EDIM 2048
#define HS   128
#define NH   16
#define TSEQ 2048
#define BATCH 4
#define BT   (BATCH * TSEQ)   // 8192

__device__ __forceinline__ v16bf join16(v8bf a, v8bf b) {
  return __builtin_shufflevector(a, b, 0,1,2,3,4,5,6,7,8,9,10,11,12,13,14,15);
}

// ---- CDNA5 async global->LDS copy (ASYNCcnt path, ISA 08_async_tensor §4) ----
__device__ __forceinline__ unsigned lds_off_of(const void* p) {
  // generic LDS pointer: addr[31:0] is the LDS byte offset (aperture in high bits)
  return (unsigned)(unsigned long long)p;
}
__device__ __forceinline__ void async_copy_b128(unsigned lds_off, const bf16* gsrc) {
  unsigned long long ga = (unsigned long long)gsrc;
  asm volatile("global_load_async_to_lds_b128 %0, %1, off"
               :: "v"(lds_off), "v"(ga) : "memory");
}
__device__ __forceinline__ void wait_async0() {
  asm volatile("s_wait_asynccnt 0" ::: "memory");
}

// ---------------------------------------------------------------- convert
__global__ void cvt_f32_to_bf16(const float* __restrict__ in,
                                bf16* __restrict__ out, int n) {
  int i = blockIdx.x * blockDim.x + threadIdx.x;
  int stride = gridDim.x * blockDim.x;
  for (; i < n; i += stride) out[i] = (bf16)in[i];
}

// ---------------------------------------------------------------- GEMM
// C[M,N] = A[M,K] (bf16, row-major) @ B[K,N] (bf16, row-major)
// block tile 128x128, 256 threads = 8 waves, wave tile 64x32, K-step 32.
template <bool OUT_F32>
__global__ __launch_bounds__(256) void gemm_bf16_wmma(
    const bf16* __restrict__ A, const bf16* __restrict__ B,
    void* __restrict__ Cout, int M, int K, int N) {
  __shared__ bf16 As[128 * 32];   // [m][k], k-stride 32
  __shared__ bf16 Bs[128 * 32];   // transposed: [n][k], k-stride 32

  const int mblk = blockIdx.x * 128;
  const int nblk = blockIdx.y * 128;
  const int tid  = threadIdx.x;
  const int wave = tid >> 5;
  const int lane = tid & 31;
  const int lo16 = lane & 15;
  const int hi   = lane >> 4;
  const int wm   = (wave & 1) * 64;    // wave m offset in block tile
  const int wn   = (wave >> 1) * 32;   // wave n offset in block tile

  v8f zero = {0.f,0.f,0.f,0.f,0.f,0.f,0.f,0.f};
  v8f acc[4][2];
#pragma unroll
  for (int i = 0; i < 4; ++i)
#pragma unroll
    for (int j = 0; j < 2; ++j) acc[i][j] = zero;

  const int arow = tid >> 1, ahalf = (tid & 1) * 16;
  const int bk = tid >> 3, bn = (tid & 7) * 16;
  const unsigned aoff0 = lds_off_of(As + arow * 32 + ahalf);
  const unsigned aoff1 = lds_off_of(As + arow * 32 + ahalf + 8);

  for (int k0 = 0; k0 < K; k0 += 32) {
    // stage A tile via async DMA to LDS (no VGPR round trip)
    {
      const bf16* src = A + (size_t)(mblk + arow) * K + k0 + ahalf;
      async_copy_b128(aoff0, src);
      async_copy_b128(aoff1, src + 8);
    }
    // stage B tile transposed -> Bs[n][k] (needs element shuffle: register path)
    {
      const bf16* src = B + (size_t)(k0 + bk) * N + nblk + bn;
#pragma unroll
      for (int i = 0; i < 16; ++i) Bs[(bn + i) * 32 + bk] = src[i];
    }
    if (k0 + 32 < K) {  // prefetch next tiles (global_prefetch_b8)
      __builtin_prefetch(A + (size_t)(mblk + arow) * K + k0 + 32 + ahalf, 0, 3);
      __builtin_prefetch(B + (size_t)(k0 + 32 + bk) * N + nblk + bn, 0, 3);
    }
    wait_async0();
    __syncthreads();

    // A fragments: lane holds row lo16, K = hi*8..+7 and 16+hi*8..+7
    v16bf af[4];
#pragma unroll
    for (int mt = 0; mt < 4; ++mt) {
      const bf16* p = As + (wm + mt * 16 + lo16) * 32 + hi * 8;
      af[mt] = join16(*(const v8bf*)p, *(const v8bf*)(p + 16));
    }
    // B fragments: lane holds col lo16, K = hi*8..+7 and 16+hi*8..+7
    v16bf bfg[2];
#pragma unroll
    for (int nt = 0; nt < 2; ++nt) {
      const bf16* p = Bs + (wn + nt * 16 + lo16) * 32 + hi * 8;
      bfg[nt] = join16(*(const v8bf*)p, *(const v8bf*)(p + 16));
    }
#pragma unroll
    for (int mt = 0; mt < 4; ++mt)
#pragma unroll
      for (int nt = 0; nt < 2; ++nt)
        acc[mt][nt] = __builtin_amdgcn_wmma_f32_16x16x32_bf16(
            false, af[mt], false, bfg[nt], (short)0, acc[mt][nt], false, false);
    __syncthreads();
  }

  // epilogue: C layout row = r + 8*hi, col = lo16
#pragma unroll
  for (int mt = 0; mt < 4; ++mt)
#pragma unroll
    for (int nt = 0; nt < 2; ++nt)
#pragma unroll
      for (int r = 0; r < 8; ++r) {
        int row = mblk + wm + mt * 16 + r + 8 * hi;
        int col = nblk + wn + nt * 16 + lo16;
        float v = acc[mt][nt][r];
        if (OUT_F32) ((float*)Cout)[(size_t)row * N + col] = v;
        else         ((bf16*)Cout)[(size_t)row * N + col] = (bf16)v;
      }
}

// ---------------------------------------------------------------- RoPE (q and v)
__global__ void rope_kernel(bf16* __restrict__ q, bf16* __restrict__ v, int total) {
  int idx = blockIdx.x * blockDim.x + threadIdx.x;
  if (idx >= total) return;                 // total = BT*NH*64
  int d  = idx & 63;
  int h  = (idx >> 6) & (NH - 1);
  int bt = idx >> 10;
  int t  = bt & (TSEQ - 1);
  float theta = __powf(10000.0f, -2.0f * (float)d / (float)HS);
  float f = (float)t * theta;
  float s, c;
  __sincosf(f, &s, &c);
  size_t base = (size_t)bt * EDIM + h * HS + d;
  float q1 = (float)q[base], q2 = (float)q[base + 64];
  q[base]      = (bf16)(q1 * c + q2 * s);
  q[base + 64] = (bf16)(-q1 * s + q2 * c);
  float v1 = (float)v[base], v2 = (float)v[base + 64];
  v[base]      = (bf16)(v1 * c + v2 * s);
  v[base + 64] = (bf16)(-v1 * s + v2 * c);
}

// ---------------------------------------------------------------- flash attention
// one workgroup per (b, h, 128-row query tile); 8 waves, 16 query rows each.
__global__ __launch_bounds__(256) void attn_kernel(
    const bf16* __restrict__ Q, const bf16* __restrict__ K,
    const bf16* __restrict__ V, bf16* __restrict__ O) {
  __shared__ bf16 Ks[32 * 128];      // [key][d]
  __shared__ bf16 Vs[128 * 32];      // transposed: [d][key]
  __shared__ bf16 Ps[8 * 16 * 32];   // per-wave P: [row][key]

  const int wg = blockIdx.x;
  const int qt = wg & 15;
  const int h  = (wg >> 4) & (NH - 1);
  const int b  = wg >> 8;
  const int tid  = threadIdx.x;
  const int wave = tid >> 5;
  const int lane = tid & 31;
  const int lo16 = lane & 15;
  const int hi   = lane >> 4;
  const int qbase = qt * 128;
  const int wrow  = qbase + wave * 16;      // wave's first query row
  const size_t headoff = (size_t)b * TSEQ * EDIM + (size_t)h * HS;
  const bf16* Qh = Q + headoff;
  const bf16* Kh = K + headoff;
  const bf16* Vh = V + headoff;

  // Q A-fragments, 4 hs-chunks of 32, loaded once from global (16B aligned)
  v16bf qf[4];
#pragma unroll
  for (int c = 0; c < 4; ++c) {
    const bf16* p = Qh + (size_t)(wrow + lo16) * EDIM + c * 32 + hi * 8;
    qf[c] = join16(*(const v8bf*)p, *(const v8bf*)(p + 16));
  }

  v8f zero = {0.f,0.f,0.f,0.f,0.f,0.f,0.f,0.f};
  v8f o[8];
#pragma unroll
  for (int i = 0; i < 8; ++i) o[i] = zero;
  float mrow[8], lrow[8];
#pragma unroll
  for (int r = 0; r < 8; ++r) { mrow[r] = -3.0e38f; lrow[r] = 0.0f; }

  const float scale = 0.08838834764831845f;  // 1/sqrt(128)
  const int ldk = tid >> 3, ldc = (tid & 7) * 16;
  const unsigned koff0 = lds_off_of(Ks + ldk * 128 + ldc);
  const unsigned koff1 = lds_off_of(Ks + ldk * 128 + ldc + 8);
  const int nkt = (qbase + 128) / 32;        // causal: key tiles needed

  for (int kt = 0; kt < nkt; ++kt) {
    const int kb = kt * 32;
    // stage K tile row-major via async DMA, V tile transposed via registers
    {
      const bf16* ksrc = Kh + (size_t)(kb + ldk) * EDIM + ldc;
      async_copy_b128(koff0, ksrc);
      async_copy_b128(koff1, ksrc + 8);
      const bf16* vsrc = Vh + (size_t)(kb + ldk) * EDIM + ldc;
#pragma unroll
      for (int i = 0; i < 16; ++i) Vs[(ldc + i) * 32 + ldk] = vsrc[i];
    }
    wait_async0();
    __syncthreads();

    // S = Q K^T : two 16-col tiles of keys
    v8f s0 = zero, s1 = zero;
#pragma unroll
    for (int c = 0; c < 4; ++c) {
      const bf16* p0 = Ks + lo16 * 128 + c * 32 + hi * 8;
      v16bf kf0 = join16(*(const v8bf*)p0, *(const v8bf*)(p0 + 16));
      s0 = __builtin_amdgcn_wmma_f32_16x16x32_bf16(false, qf[c], false, kf0,
                                                   (short)0, s0, false, false);
      const bf16* p1 = Ks + (16 + lo16) * 128 + c * 32 + hi * 8;
      v16bf kf1 = join16(*(const v8bf*)p1, *(const v8bf*)(p1 + 16));
      s1 = __builtin_amdgcn_wmma_f32_16x16x32_bf16(false, qf[c], false, kf1,
                                                   (short)0, s1, false, false);
    }

    // online softmax per row (row = wrow + r + 8*hi, reduce over 16-lane N group)
#pragma unroll
    for (int r = 0; r < 8; ++r) {
      const int row = wrow + r + 8 * hi;
      float a = s0[r] * scale; if (kb + lo16 > row)      a = -1e30f;
      float bb = s1[r] * scale; if (kb + 16 + lo16 > row) bb = -1e30f;
      float mx = fmaxf(a, bb);
      mx = fmaxf(mx, __shfl_xor(mx, 1));
      mx = fmaxf(mx, __shfl_xor(mx, 2));
      mx = fmaxf(mx, __shfl_xor(mx, 4));
      mx = fmaxf(mx, __shfl_xor(mx, 8));
      const float mnew  = fmaxf(mrow[r], mx);
      const float alpha = __expf(mrow[r] - mnew);
      const float p0 = __expf(a - mnew);
      const float p1 = __expf(bb - mnew);
      float ps = p0 + p1;
      ps += __shfl_xor(ps, 1);
      ps += __shfl_xor(ps, 2);
      ps += __shfl_xor(ps, 4);
      ps += __shfl_xor(ps, 8);
      lrow[r] = lrow[r] * alpha + ps;
      mrow[r] = mnew;
#pragma unroll
      for (int nt = 0; nt < 8; ++nt) o[nt][r] *= alpha;
      bf16* pp = Ps + wave * (16 * 32) + (r + 8 * hi) * 32;
      pp[lo16]      = (bf16)p0;
      pp[16 + lo16] = (bf16)p1;
    }
    asm volatile("s_wait_dscnt 0" ::: "memory");   // wave-local LDS RAW

    // P A-fragment (16x32) from per-wave LDS
    const bf16* pf = Ps + wave * (16 * 32) + lo16 * 32 + hi * 8;
    v16bf pfr = join16(*(const v8bf*)pf, *(const v8bf*)(pf + 16));

    // O += P @ V : 8 hs tiles of 16
#pragma unroll
    for (int nt = 0; nt < 8; ++nt) {
      const bf16* vp = Vs + (nt * 16 + lo16) * 32 + hi * 8;
      v16bf vf = join16(*(const v8bf*)vp, *(const v8bf*)(vp + 16));
      o[nt] = __builtin_amdgcn_wmma_f32_16x16x32_bf16(false, pfr, false, vf,
                                                      (short)0, o[nt], false, false);
    }
    __syncthreads();
  }

  // normalize and write context (bf16, [b,t,h*128+d] layout)
  bf16* Oh = O + headoff;
#pragma unroll
  for (int r = 0; r < 8; ++r) {
    const int row = wrow + r + 8 * hi;
    const float inv = 1.0f / lrow[r];
#pragma unroll
    for (int nt = 0; nt < 8; ++nt)
      Oh[(size_t)row * EDIM + nt * 16 + lo16] = (bf16)(o[nt][r] * inv);
  }
}

// ---------------------------------------------------------------- launcher
extern "C" void kernel_launch(void* const* d_in, const int* in_sizes, int n_in,
                              void* d_out, int out_size, void* d_ws, size_t ws_size,
                              hipStream_t stream) {
  const float* x  = (const float*)d_in[0];
  const float* Wq = (const float*)d_in[1];
  const float* Wk = (const float*)d_in[2];
  const float* Wv = (const float*)d_in[3];
  const float* Wo = (const float*)d_in[4];
  float* out = (float*)d_out;

  const size_t nX = (size_t)BT * EDIM;        // 16,777,216
  const size_t nW = (size_t)EDIM * EDIM;      // 4,194,304

  bf16* ws = (bf16*)d_ws;
  bf16* xb  = ws;            ws += nX;
  bf16* wqb = ws;            ws += nW;
  bf16* wkb = ws;            ws += nW;
  bf16* wvb = ws;            ws += nW;
  bf16* wob = ws;            ws += nW;
  bf16* Qb  = ws;            ws += nX;
  bf16* Kb  = ws;            ws += nX;
  bf16* Vb  = ws;            ws += nX;
  bf16* Cb  = ws;            ws += nX;

  // 1) convert inputs to bf16
  cvt_f32_to_bf16<<<4096, 256, 0, stream>>>(x,  xb,  (int)nX);
  cvt_f32_to_bf16<<<4096, 256, 0, stream>>>(Wq, wqb, (int)nW);
  cvt_f32_to_bf16<<<4096, 256, 0, stream>>>(Wk, wkb, (int)nW);
  cvt_f32_to_bf16<<<4096, 256, 0, stream>>>(Wv, wvb, (int)nW);
  cvt_f32_to_bf16<<<4096, 256, 0, stream>>>(Wo, wob, (int)nW);

  // 2) QKV projections
  dim3 gg(BT / 128, EDIM / 128);
  gemm_bf16_wmma<false><<<gg, 256, 0, stream>>>(xb, wqb, Qb, BT, EDIM, EDIM);
  gemm_bf16_wmma<false><<<gg, 256, 0, stream>>>(xb, wkb, Kb, BT, EDIM, EDIM);
  gemm_bf16_wmma<false><<<gg, 256, 0, stream>>>(xb, wvb, Vb, BT, EDIM, EDIM);

  // 3) RoPE on Q and V (matches reference)
  const int ropeN = BT * NH * 64;
  rope_kernel<<<(ropeN + 255) / 256, 256, 0, stream>>>(Qb, Vb, ropeN);

  // 4) causal flash attention -> context
  attn_kernel<<<BATCH * NH * (TSEQ / 128), 256, 0, stream>>>(Qb, Kb, Vb, Cb);

  // 5) output projection -> f32
  gemm_bf16_wmma<true><<<gg, 256, 0, stream>>>(Cb, wob, out, BT, EDIM, EDIM);
}
